// retrieval_10926396801549
// MI455X (gfx1250) — matrix operand order repaired and checked by reading, not compiled, not run on hardware
//
#include <hip/hip_runtime.h>
#include <hip/hip_bf16.h>

typedef __attribute__((ext_vector_type(16))) __bf16 v16bf;
typedef __attribute__((ext_vector_type(8)))  float  v8f;

#define N_Q   256       // queries
#define DIM   768       // feature dim
#define KT    24        // K tiles of 32 (768/32)
#define N_DB  131072    // database rows
#define L_Y   512       // payload row length
#define ROWS_PER_CHUNK 16
#define ROWGRPS 256     // row-range workgroups
#define CHUNKS ((N_DB / ROWS_PER_CHUNK) / ROWGRPS)   // 32 chunks of 16 rows = 512 rows/WG

__device__ __forceinline__ unsigned short f32_to_bf16_rne(float f) {
    unsigned u = __float_as_uint(f);
    unsigned r = u + 0x7FFFu + ((u >> 16) & 1u);   // round-to-nearest-even
    return (unsigned short)(r >> 16);
}

// monotone float -> uint map (total order, preserves <)
__device__ __forceinline__ unsigned f2ord(float f) {
    unsigned u = __float_as_uint(f);
    return (u & 0x80000000u) ? ~u : (u | 0x80000000u);
}

__global__ void init_keys(unsigned long long* keys) {
    keys[threadIdx.x] = 0ull;
}

// grid = dim3(2, ROWGRPS): blockIdx.x = M-half (fast-varying so paired blocks
// stream the same DB rows and reuse them through the 192MB L2),
// blockIdx.y = row range. block = 256 threads = 8 waves; wave w owns M-tile
// (queries 16*(8*mhalf + w) .. +15).
__global__ __launch_bounds__(256)
__attribute__((amdgpu_waves_per_eu(4)))
void topk_gemm(const float* __restrict__ imu,
               const float* __restrict__ dbx,
               unsigned long long* __restrict__ keys) {
    // chunk staged in LDS, pre-swizzled into WMMA B-fragment order:
    // [kt 0..23][lane 0..31][slot 0..15] bf16 ; lane = 16*(kin/16) + row, slot = kin%16
    __shared__ __align__(32) unsigned short lds[KT * 32 * 16];   // 24 KB
    __shared__ float lds_rs[ROWS_PER_CHUNK];                     // per-row 1/norm

    const int tid   = threadIdx.x;
    const int wave  = tid >> 5;
    const int lane  = tid & 31;
    const int half  = lane >> 4;
    const int l15   = lane & 15;
    const int mhalf = blockIdx.x;

    // ---- A panel: this wave's 16 queries, bf16, contiguous k order (k = 16*half + e) ----
    // lane L (L<16): row M=L, VGPR0..7 hold K=0..15
    // lane L (L>=16): same rows, K=16..31
    v16bf a[KT];
    {
        const int q = ((mhalf * 8 + wave) << 4) + l15;
        const float* __restrict__ qrow = imu + (size_t)q * DIM;
        #pragma unroll
        for (int t = 0; t < KT; ++t) {
            union { v16bf v; unsigned short s[16]; } u;
            #pragma unroll
            for (int e = 0; e < 16; ++e) {
                const int k = t * 32 + half * 16 + e;
                u.s[e] = f32_to_bf16_rne(qrow[k]);
            }
            a[t] = u.v;
        }
    }

    float    bestv[8];
    #pragma unroll
    for (int v = 0; v < 8; ++v) bestv[v] = -3.0e38f;
    unsigned bc_lo = 0u, bc_hi = 0u;   // packed best-chunk id (1 byte per v)

    // i-invariant part of the LDS scatter address (ushort index):
    // full index = fragbase + (r<<4) + i*2048  (i.e. byte offset i*4096)
    const int fragbase = ((lane >> 3) << 9) + (((lane >> 2) & 1) << 8) + ((lane & 3) << 2);

    const int wg_row0 = blockIdx.y * (CHUNKS * ROWS_PER_CHUNK);

    #pragma unroll 1   // one iteration in flight: keeps the A panel resident
    for (int c = 0; c < CHUNKS; ++c) {
        const int row0 = wg_row0 + c * ROWS_PER_CHUNK;

        // ---- stream one chunk: each wave handles 2 rows; store raw bf16,
        //      normalization deferred to the fp32 accumulator ----
        #pragma unroll
        for (int rr = 0; rr < 2; ++rr) {
            const int r = wave * 2 + rr;                 // 0..15
            const float* __restrict__ drow = dbx + (size_t)(row0 + r) * DIM;
            unsigned short* __restrict__ dstp = &lds[fragbase + (r << 4)];
            float ss = 0.f;
            // partial unroll: bursts of 3 float4 loads -> lower transient VGPR
            // peak (these mostly hit L2 thanks to the prefetch a chunk earlier)
            #pragma unroll 3
            for (int i = 0; i < 6; ++i) {                // coalesced b128 loads
                const float4 f = *(const float4*)(drow + i * 128 + lane * 4);
                ss += f.x*f.x + f.y*f.y + f.z*f.z + f.w*f.w;
                const unsigned lo = (unsigned)f32_to_bf16_rne(f.x) |
                                    ((unsigned)f32_to_bf16_rne(f.y) << 16);
                const unsigned hi = (unsigned)f32_to_bf16_rne(f.z) |
                                    ((unsigned)f32_to_bf16_rne(f.w) << 16);
                // one ds_store_b64 per float4, immediate offset i*4096 bytes
                *(uint2*)(dstp + i * 2048) = make_uint2(lo, hi);
            }
            #pragma unroll
            for (int m = 16; m >= 1; m >>= 1) ss += __shfl_xor(ss, m, 32);
            if (lane == 0) lds_rs[r] = 1.0f / fmaxf(sqrtf(ss), 1e-8f);
        }
        __syncthreads();

        // prefetch next chunk while the matrix units work (global_prefetch_b8)
        if (c + 1 < CHUNKS) {
            const char* nxt = (const char*)(dbx + (size_t)(row0 + ROWS_PER_CHUNK) * DIM);
            __builtin_prefetch(nxt + (size_t)tid * 128, 0, 3);
            if (tid < 128) __builtin_prefetch(nxt + (size_t)(tid + 256) * 128, 0, 3);
        }

        // ---- 16x16 tile of sims via 24 chained bf16 WMMAs,
        //      B fragments software-pipelined 1 tile ahead ----
        v8f acc = {};
        v16bf bcur = *(const v16bf*)&lds[(0 * 32 + lane) << 4];  // 2x ds_load_b128
        #pragma unroll
        for (int t = 0; t < KT - 1; ++t) {
            const v16bf bnxt = *(const v16bf*)&lds[((t + 1) * 32 + lane) << 4];
            acc = __builtin_amdgcn_wmma_f32_16x16x32_bf16(
                      false, a[t], false, bcur, (short)0, acc, false, false);
            bcur = bnxt;
        }
        acc = __builtin_amdgcn_wmma_f32_16x16x32_bf16(
                  false, a[KT - 1], false, bcur, (short)0, acc, false, false);

        // C/D layout: M = v + 8*half, N = l15 -> this lane always sees row row0+l15
        const float rsv = lds_rs[l15];                   // per-row 1/norm
        #pragma unroll
        for (int v = 0; v < 8; ++v) {
            const float s = acc[v] * rsv;
            if (s > bestv[v]) {                          // strict >: keep smallest idx
                bestv[v] = s;
                const int sh = (v & 3) << 3;
                if (v < 4) bc_lo = (bc_lo & ~(0xffu << sh)) | ((unsigned)c << sh);
                else       bc_hi = (bc_hi & ~(0xffu << sh)) | ((unsigned)c << sh);
            }
        }
        __syncthreads();
    }

    // ---- argmax across the 16 lanes of each half (N dimension) ----
    #pragma unroll
    for (int v = 0; v < 8; ++v) {
        const unsigned cb = ((v < 4 ? bc_lo : bc_hi) >> ((v & 3) << 3)) & 0xffu;
        float    bv = bestv[v];
        unsigned bi = (unsigned)(wg_row0 + ((int)cb << 4) + l15);
        #pragma unroll
        for (int m = 1; m <= 8; m <<= 1) {
            const float    ov = __shfl_xor(bv, m, 32);
            const unsigned oi = __shfl_xor(bi, m, 32);
            if (ov > bv || (ov == bv && oi < bi)) { bv = ov; bi = oi; }
        }
        if (l15 == 0) {
            const int q = ((mhalf * 8 + wave) << 4) + v + (half << 3);
            // pack (ordered sim, ~idx): atomicMax picks larger sim, ties -> smaller idx
            const unsigned long long key =
                ((unsigned long long)f2ord(bv) << 32) | (unsigned)(~bi);
            atomicMax(&keys[q], key);
        }
    }
}

__global__ void gather_y(const unsigned long long* __restrict__ keys,
                         const float* __restrict__ dby,
                         float* __restrict__ out) {
    const int q = blockIdx.x;
    const unsigned idx = ~(unsigned)(keys[q] & 0xFFFFFFFFull);
    const float4* __restrict__ src = (const float4*)(dby + (size_t)idx * L_Y);
    float4* __restrict__ dst = (float4*)(out + (size_t)q * L_Y);
    dst[threadIdx.x] = src[threadIdx.x];   // 128 threads x float4 = 512 floats
}

extern "C" void kernel_launch(void* const* d_in, const int* in_sizes, int n_in,
                              void* d_out, int out_size, void* d_ws, size_t ws_size,
                              hipStream_t stream) {
    const float* imu = (const float*)d_in[0];   // [256, 6, 128] = [256, 768]
    const float* dbx = (const float*)d_in[1];   // [131072, 768]
    const float* dby = (const float*)d_in[2];   // [131072, 512]
    float* out = (float*)d_out;                 // [256, 512]
    unsigned long long* keys = (unsigned long long*)d_ws;   // 256 * 8B

    init_keys<<<1, N_Q, 0, stream>>>(keys);
    topk_gemm<<<dim3(2, ROWGRPS), 256, 0, stream>>>(imu, dbx, keys);
    gather_y<<<N_Q, L_Y / 4, 0, stream>>>(keys, dby, out);
}
